// LSTM_33380485824995
// MI455X (gfx1250) — matrix operand order repaired and checked by reading, not compiled, not run on hardware
//
#include <hip/hip_runtime.h>

typedef __attribute__((ext_vector_type(16))) _Float16 v16h;
typedef __attribute__((ext_vector_type(8)))  _Float16 v8h;
typedef __attribute__((ext_vector_type(8)))  float    v8f;

#define IDIM 32
#define HDIM 256
#define CDIM 16
#define BDIM 64
#define TDIM 2048
#define GDIM 1024  // 4*H

// ---------------- device helpers ----------------

__device__ __forceinline__ float fast_rcp(float x) {
  return __builtin_amdgcn_rcpf(x);   // v_rcp_f32 (TRANS32, co-executes with WMMA)
}
__device__ __forceinline__ float sig_f(float x) {
  return fast_rcp(1.0f + __expf(-x));
}
__device__ __forceinline__ float tanh_f(float x) {
  // tanh(x) = 1 - 2/(exp(2x)+1); saturates correctly at +/-1
  float e = __expf(2.0f * x);
  return 1.0f - 2.0f * fast_rcp(e + 1.0f);
}

__device__ __forceinline__ v8f wmma16(v16h a, v16h b, v8f c) {
  // (neg_a, A, neg_b, B, c_mod, C, reuse_a, reuse_b)
  return __builtin_amdgcn_wmma_f32_16x16x32_f16(false, a, false, b, (short)0, c,
                                                false, false);
}

// A-fragment (16x32 f16) from a row-major f16 buffer (rowStride in elements).
// ISA layout: lanes 0-15: M=lane, K runs [kt*32+0..7] and [kt*32+16..23];
//             lanes 16-31: M=lane-16, K runs [+8..15] and [+24..31].
__device__ __forceinline__ v16h load_A(const _Float16* base, int rowStride,
                                       int kt, int lane) {
  const int m   = lane & 15;
  const int ofs = (lane & 16) ? 8 : 0;
  const _Float16* row = base + m * rowStride + kt * 32;
  v8h lo = *(const v8h*)(row + ofs);
  v8h hi = *(const v8h*)(row + 16 + ofs);
  union { v16h v; v8h h[2]; } u;
  u.h[0] = lo; u.h[1] = hi;
  return u.v;
}

// B-fragment (32x16 f16) from pre-packed weights: [nt][kt][lane][16] f16.
__device__ __forceinline__ v16h load_B(const _Float16* pw, int nt, int kt,
                                       int KT, int lane) {
  const _Float16* p = pw + (((size_t)nt * KT + kt) * 32 + lane) * 16;
  return *(const v16h*)p;
}

// ---------------- pre-pass kernels ----------------

// Pack W [N][K] (row-major f32, used as B[k][n] = W[n][k]) into WMMA B-fragment
// order: pw[((nt*KT + kt)*32 + lane)*16 + e] = W[nt*16+(lane&15)][kt*32+((lane&16)>>1)+(e&7)+((e&8)<<1)]
__global__ void lstm_pack_B(const float* __restrict__ W,
                            _Float16* __restrict__ pw, int N, int K) {
  int idx = blockIdx.x * blockDim.x + threadIdx.x;
  int total = N * K;
  if (idx >= total) return;
  int e    = idx & 15;
  int lane = (idx >> 4) & 31;
  int rest = idx >> 9;
  int KT = K >> 5;
  int kt = rest % KT;
  int nt = rest / KT;
  int n = nt * 16 + (lane & 15);
  int k = kt * 32 + ((lane & 16) >> 1) + (e & 7) + ((e & 8) << 1);
  pw[idx] = (_Float16)W[(size_t)n * K + k];
}

__global__ void lstm_combine_bias(const float* __restrict__ a,
                                  const float* __restrict__ b,
                                  float* __restrict__ o, int n) {
  int i = blockIdx.x * blockDim.x + threadIdx.x;
  if (i < n) o[i] = a[i] + b[i];
}

// ---------------- fused persistent 2-layer LSTM + FC ----------------
// grid.x = 4 (batch tiles of 16), block = 512 threads = 16 waves.
// Wave w owns gate tiles {w, 16+w, 32+w, 48+w} = {i,f,g,o} for h-cols [16w,16w+16).
// Cell state c0/c1 lives in registers (f32); h0/h1 live in LDS (f16).
__launch_bounds__(512, 1)
__global__ void lstm_fused(const float* __restrict__ x,      // [B,T,I]
                           const _Float16* __restrict__ pWih0,   // KT=1
                           const _Float16* __restrict__ pWhh0,   // KT=8
                           const _Float16* __restrict__ pWih1,   // KT=8
                           const _Float16* __restrict__ pWhh1,   // KT=8
                           const _Float16* __restrict__ pFc,     // KT=8, 1 ntile
                           const float* __restrict__ bias0,      // [1024]
                           const float* __restrict__ bias1,      // [1024]
                           const float* __restrict__ fcb,        // [16]
                           float* __restrict__ out)              // [B,T,C]
{
  __shared__ _Float16 h0[16][HDIM];   // 8 KB
  __shared__ _Float16 h1[16][HDIM];   // 8 KB
  __shared__ _Float16 xin[16][IDIM];  // 1 KB

  const int tid    = threadIdx.x;
  const int w      = tid >> 5;        // wave 0..15
  const int lane   = tid & 31;
  const int ncol   = lane & 15;
  const int rowoff = (lane & 16) ? 8 : 0;
  const int b0     = blockIdx.x * 16;

  // loop-invariant biases -> registers (per wave / per lane)
  float bv0[4], bv1[4];
#pragma unroll
  for (int g = 0; g < 4; ++g) {
    bv0[g] = bias0[(g * 16 + w) * 16 + ncol];
    bv1[g] = bias1[(g * 16 + w) * 16 + ncol];
  }
  const float fcbias = fcb[ncol];

  // zero-init recurrent state
  for (int idx = tid; idx < 16 * HDIM; idx += 512) {
    int m = idx >> 8, j = idx & 255;
    h0[m][j] = (_Float16)0.0f;
    h1[m][j] = (_Float16)0.0f;
  }
  float c0frag[8], c1frag[8];
#pragma unroll
  for (int r = 0; r < 8; ++r) { c0frag[r] = 0.0f; c1frag[r] = 0.0f; }
  __syncthreads();

  for (int t = 0; t < TDIM; ++t) {
    // ---- stage x_t tile [16 x 32] f32 -> f16 LDS ----
    {
      int m = tid >> 5, i = tid & 31;   // 512 threads, one element each
      xin[m][i] = (_Float16)x[((size_t)(b0 + m) * TDIM + t) * IDIM + i];
    }
    __syncthreads();

    // ================= layer 0 =================
    v8f acc[4];
#pragma unroll
    for (int g = 0; g < 4; ++g) {
#pragma unroll
      for (int r = 0; r < 8; ++r) acc[g][r] = bv0[g];
    }
    {  // input contribution, K = 32
      v16h a = load_A(&xin[0][0], IDIM, 0, lane);
#pragma unroll
      for (int g = 0; g < 4; ++g)
        acc[g] = wmma16(a, load_B(pWih0, g * 16 + w, 0, 1, lane), acc[g]);
    }
#pragma unroll 1
    for (int kt = 0; kt < 8; ++kt) {  // recurrent, K = 256
      v16h a = load_A(&h0[0][0], HDIM, kt, lane);
#pragma unroll
      for (int g = 0; g < 4; ++g)
        acc[g] = wmma16(a, load_B(pWhh0, g * 16 + w, kt, 8, lane), acc[g]);
    }
    // gate nonlinearity + cell update entirely in registers
#pragma unroll
    for (int r = 0; r < 8; ++r) {
      float c = sig_f(acc[1][r]) * c0frag[r] + sig_f(acc[0][r]) * tanh_f(acc[2][r]);
      c0frag[r] = c;
      float hv = sig_f(acc[3][r]) * tanh_f(c);
      h0[r + rowoff][w * 16 + ncol] = (_Float16)hv;
    }
    __syncthreads();

    // ================= layer 1 =================
#pragma unroll
    for (int g = 0; g < 4; ++g) {
#pragma unroll
      for (int r = 0; r < 8; ++r) acc[g][r] = bv1[g];
    }
#pragma unroll 1
    for (int kt = 0; kt < 8; ++kt) {  // input contribution from h0, K = 256
      v16h a = load_A(&h0[0][0], HDIM, kt, lane);
#pragma unroll
      for (int g = 0; g < 4; ++g)
        acc[g] = wmma16(a, load_B(pWih1, g * 16 + w, kt, 8, lane), acc[g]);
    }
#pragma unroll 1
    for (int kt = 0; kt < 8; ++kt) {  // recurrent from h1, K = 256
      v16h a = load_A(&h1[0][0], HDIM, kt, lane);
#pragma unroll
      for (int g = 0; g < 4; ++g)
        acc[g] = wmma16(a, load_B(pWhh1, g * 16 + w, kt, 8, lane), acc[g]);
    }
#pragma unroll
    for (int r = 0; r < 8; ++r) {
      float c = sig_f(acc[1][r]) * c1frag[r] + sig_f(acc[0][r]) * tanh_f(acc[2][r]);
      c1frag[r] = c;
      float hv = sig_f(acc[3][r]) * tanh_f(c);
      h1[r + rowoff][w * 16 + ncol] = (_Float16)hv;
    }
    __syncthreads();

    // ================= FC head (wave 0): out = h1 @ fcW^T + fcb =========
    if (w == 0) {
      v8f o;
#pragma unroll
      for (int r = 0; r < 8; ++r) o[r] = fcbias;
#pragma unroll 1
      for (int kt = 0; kt < 8; ++kt)
        o = wmma16(load_A(&h1[0][0], HDIM, kt, lane),
                   load_B(pFc, 0, kt, 8, lane), o);
#pragma unroll
      for (int r = 0; r < 8; ++r)
        out[((size_t)(b0 + r + rowoff) * TDIM + t) * CDIM + ncol] = o[r];
    }
    // next h1 write is 2 barriers away, so FC read is safe without extra sync
  }
}

// ---------------- host launch ----------------

extern "C" void kernel_launch(void* const* d_in, const int* in_sizes, int n_in,
                              void* d_out, int out_size, void* d_ws, size_t ws_size,
                              hipStream_t stream) {
  const float* seq   = (const float*)d_in[0];   // [B,T,I]
  const float* Wih0  = (const float*)d_in[1];   // [4H,I]
  const float* Whh0  = (const float*)d_in[2];   // [4H,H]
  const float* bih0  = (const float*)d_in[3];
  const float* bhh0  = (const float*)d_in[4];
  const float* Wih1  = (const float*)d_in[5];   // [4H,H]
  const float* Whh1  = (const float*)d_in[6];   // [4H,H]
  const float* bih1  = (const float*)d_in[7];
  const float* bhh1  = (const float*)d_in[8];
  const float* fcW   = (const float*)d_in[9];   // [C,H]
  const float* fcb   = (const float*)d_in[10];
  float* out = (float*)d_out;

  // workspace layout (bytes)
  char* ws = (char*)d_ws;
  _Float16* pWih0 = (_Float16*)(ws + 0);                 //  64 KB (1024x32)
  _Float16* pWhh0 = (_Float16*)(ws + 65536);             // 512 KB (1024x256)
  _Float16* pWih1 = (_Float16*)(ws + 65536 + 524288);
  _Float16* pWhh1 = (_Float16*)(ws + 65536 + 2 * 524288);
  _Float16* pFc   = (_Float16*)(ws + 65536 + 3 * 524288);            // 8 KB (16x256)
  float*    bias0 = (float*)   (ws + 65536 + 3 * 524288 + 8192);
  float*    bias1 = (float*)   (ws + 65536 + 3 * 524288 + 8192 + 4096);

  // pack weights into WMMA B-fragment order (f32 -> f16)
  lstm_pack_B<<<(GDIM * IDIM + 255) / 256, 256, 0, stream>>>(Wih0, pWih0, GDIM, IDIM);
  lstm_pack_B<<<(GDIM * HDIM + 255) / 256, 256, 0, stream>>>(Whh0, pWhh0, GDIM, HDIM);
  lstm_pack_B<<<(GDIM * HDIM + 255) / 256, 256, 0, stream>>>(Wih1, pWih1, GDIM, HDIM);
  lstm_pack_B<<<(GDIM * HDIM + 255) / 256, 256, 0, stream>>>(Whh1, pWhh1, GDIM, HDIM);
  lstm_pack_B<<<(CDIM * HDIM + 255) / 256, 256, 0, stream>>>(fcW, pFc, CDIM, HDIM);
  lstm_combine_bias<<<(GDIM + 255) / 256, 256, 0, stream>>>(bih0, bhh0, bias0, GDIM);
  lstm_combine_bias<<<(GDIM + 255) / 256, 256, 0, stream>>>(bih1, bhh1, bias1, GDIM);

  // persistent fused kernel: 4 workgroups (one per 16-row batch tile)
  lstm_fused<<<BDIM / 16, 512, 0, stream>>>(seq, pWih0, pWhh0, pWih1, pWhh1, pFc,
                                            bias0, bias1, fcb, out);
}
